// Attention_88467736363359
// MI455X (gfx1250) — matrix operand order repaired and checked by reading, not compiled, not run on hardware
//
#include <hip/hip_runtime.h>
#include <hip/hip_bf16.h>
#include <stdint.h>

// ---------------------------------------------------------------------------
// Attention (B=2, N=2048, DIM=1024, H=16, DH=64) for gfx1250 (MI455X).
// bf16 WMMA (v_wmma_f32_16x16x32_bf16); f32 softmax; double-buffered GEMMs.
// Wave-uniform scalarization (readfirstlane) + u32 offsets -> saddr+voffset
// addressing, no per-lane 64-bit pointer math, no spills.
// ---------------------------------------------------------------------------

#define BATCH  2
#define SEQ    2048
#define DIMM   1024
#define DH     64
#define HEADS  16
#define PBD    256
#define NREL   (2 * SEQ - 1)   // 4095

typedef __bf16 bf16_t;
typedef __attribute__((ext_vector_type(16))) __bf16 v16bf;
typedef __attribute__((ext_vector_type(8)))  float  v8f;

union Frag {            // 32 bytes: one wave-lane's share of a 16x32 bf16 tile
    v16bf v;
    uint4 q[2];
};

__device__ __forceinline__ v8f wmma_bf16f32(v16bf a, v16bf b, v8f c) {
    return __builtin_amdgcn_wmma_f32_16x16x32_bf16(
        /*neg_a=*/false, a, /*neg_b=*/false, b,
        /*c_mod=*/(short)0, c, /*reuse_a=*/false, /*reuse_b=*/false);
}

__device__ __forceinline__ float siluf(float x) {
    return x * (1.0f / (1.0f + __expf(-x)));
}

// 16-byte load at a 32-bit element offset from a wave-uniform base pointer.
__device__ __forceinline__ uint4 ldg4(const bf16_t* __restrict__ base, unsigned elt) {
    return *reinterpret_cast<const uint4*>(base + elt);
}
// A-fragment: 16x32 bf16; lane m=lane&15, K halves g*8+{0..7} and g*8+16+{0..7}
__device__ __forceinline__ void load_afrag(Frag& f, const bf16_t* __restrict__ base, unsigned elt) {
    f.q[0] = ldg4(base, elt);
    f.q[1] = ldg4(base, elt + 16);
}
// B-fragment: 32x16 bf16 from [N][K]-major storage; K halves g*8+{0..7} and g*8+16+{0..7}
__device__ __forceinline__ void load_bfrag(Frag& f, const bf16_t* __restrict__ base, unsigned elt) {
    f.q[0] = ldg4(base, elt);
    f.q[1] = ldg4(base, elt + 16);
}

// ---------------------------------------------------------------------------
// 1) RMSNorm: xn = x / max(||x||,1e-12) * sqrt(DIM) * g   -> bf16
// ---------------------------------------------------------------------------
__global__ void rmsnorm_kernel(const float* __restrict__ x,
                               const float* __restrict__ g,
                               bf16_t* __restrict__ xn) {
    const int row = blockIdx.x;
    const int tid = threadIdx.x;
    const float4 xv = reinterpret_cast<const float4*>(x + (size_t)row * DIMM)[tid];
    float ss = xv.x * xv.x + xv.y * xv.y + xv.z * xv.z + xv.w * xv.w;
#pragma unroll
    for (int m = 1; m < 32; m <<= 1) ss += __shfl_xor(ss, m, 32);
    __shared__ float red[8];
    if ((tid & 31) == 0) red[tid >> 5] = ss;
    __syncthreads();
    float tot = 0.0f;
#pragma unroll
    for (int i = 0; i < 8; ++i) tot += red[i];
    const float rinv = 32.0f / fmaxf(sqrtf(tot), 1e-12f);   // sqrt(1024)=32
    const float4 gv = reinterpret_cast<const float4*>(g)[tid];
    bf16_t* o = xn + (size_t)row * DIMM + tid * 4;
    o[0] = (bf16_t)(xv.x * rinv * gv.x);
    o[1] = (bf16_t)(xv.y * rinv * gv.y);
    o[2] = (bf16_t)(xv.z * rinv * gv.z);
    o[3] = (bf16_t)(xv.w * rinv * gv.w);
}

// ---------------------------------------------------------------------------
// 2) Weight transpose + f32->bf16:  wT[n][k] = w[k][n]
// ---------------------------------------------------------------------------
__global__ void conv_transpose_kernel(const float* __restrict__ w,
                                      bf16_t* __restrict__ wT,
                                      int K, int Ncols) {
    const unsigned idx = blockIdx.x * blockDim.x + threadIdx.x;
    const unsigned n = idx / K;
    const unsigned k = idx % K;
    wT[idx] = (bf16_t)w[(size_t)k * Ncols + n];
}

// ---------------------------------------------------------------------------
// 3) DynamicPositionBias MLP -> pb_t[4095][16]
// ---------------------------------------------------------------------------
__global__ void pb_mlp_kernel(const float* __restrict__ w1, const float* __restrict__ b1,
                              const float* __restrict__ w2, const float* __restrict__ b2,
                              const float* __restrict__ w3, const float* __restrict__ b3,
                              float* __restrict__ pb_t) {
    const int row = blockIdx.x;                 // 0..4094
    const int j = threadIdx.x;                  // 0..255
    const float pos = (float)(row - (SEQ - 1)); // -2047..2047
    __shared__ float h1[PBD];
    __shared__ float h2[PBD];
    h1[j] = siluf(pos * w1[j] + b1[j]);
    __syncthreads();
    float acc = b2[j];
    for (int t = 0; t < PBD; ++t) acc += h1[t] * w2[t * PBD + j];
    h2[j] = siluf(acc);
    __syncthreads();
    if (j < HEADS) {
        float o = b3[j];
        for (int t = 0; t < PBD; ++t) o += h2[t] * w3[t * HEADS + j];
        pb_t[row * HEADS + j] = o;
    }
}

// ---------------------------------------------------------------------------
// 4) QKV GEMM: [4096,1024]bf16 x [1024,3072]bf16.  Wave tile 32(M)x64(N),
//    K in steps of 32, double-buffered fragments (8 wmma / iteration).
//    Q pre-scaled by 0.125; V written transposed [b,h,64,n].
// ---------------------------------------------------------------------------
__global__ void qkv_gemm_kernel(const bf16_t* __restrict__ A,
                                const bf16_t* __restrict__ Bt,   // [3072][1024]
                                bf16_t* __restrict__ Qb,
                                bf16_t* __restrict__ Kb,
                                bf16_t* __restrict__ Vt) {
    const int wid = __builtin_amdgcn_readfirstlane(
        blockIdx.x * (blockDim.x >> 5) + (threadIdx.x >> 5));
    const int lane = threadIdx.x & 31;
    const int ml = lane & 15, gr = lane >> 4;
    const int mt = wid & 127;        // 128 M-tiles of 32 rows
    const int nb = wid >> 7;         // 48 N-tiles of 64 cols

    // u32 element offsets off wave-uniform bases (everything < 2^22 elements)
    const unsigned aoff0 = (unsigned)(mt * 32 + ml) * DIMM + gr * 8;
    const unsigned aoff1 = aoff0 + 16u * DIMM;
    unsigned boff[4];
#pragma unroll
    for (int nd = 0; nd < 4; ++nd)
        boff[nd] = (unsigned)(nb * 64 + nd * 16 + ml) * DIMM + gr * 8;

    v8f acc[2][4] = {};
    Frag af[2][2];                   // [buffer][m-subtile]
    Frag bfm[2][4];                  // [buffer][n-subtile]
    load_afrag(af[0][0], A, aoff0);
    load_afrag(af[0][1], A, aoff1);
#pragma unroll
    for (int nd = 0; nd < 4; ++nd) load_bfrag(bfm[0][nd], Bt, boff[nd]);

    for (int kk = 0; kk < DIMM; kk += 32) {
        const int cur = (kk >> 5) & 1;
        const int nxt = cur ^ 1;
        if (kk + 32 < DIMM) {        // prefetch next K-slice into alt buffer
            load_afrag(af[nxt][0], A, aoff0 + kk + 32);
            load_afrag(af[nxt][1], A, aoff1 + kk + 32);
#pragma unroll
            for (int nd = 0; nd < 4; ++nd) load_bfrag(bfm[nxt][nd], Bt, boff[nd] + kk + 32);
        }
#pragma unroll
        for (int s = 0; s < 2; ++s)
#pragma unroll
            for (int nd = 0; nd < 4; ++nd)
                acc[s][nd] = wmma_bf16f32(af[cur][s].v, bfm[cur][nd].v, acc[s][nd]);
    }

    // Epilogue: branch on nb (wave-uniform): nb<16 -> Q, <32 -> K, else V.
    const int colbase = nb * 64;
#pragma unroll
    for (int s = 0; s < 2; ++s) {
#pragma unroll
        for (int nd = 0; nd < 4; ++nd) {
            const int col = colbase + nd * 16 + ml;
#pragma unroll
            for (int r = 0; r < 8; ++r) {
                const int row = mt * 32 + s * 16 + r + 8 * gr;
                const unsigned b = (unsigned)row >> 11;       // /2048
                const unsigned i = (unsigned)row & (SEQ - 1);
                const float val = acc[s][nd][r];
                if (nb < 16) {
                    const unsigned c2 = (unsigned)col;
                    const unsigned h = c2 >> 6, dh = c2 & 63;
                    Qb[(((b * HEADS + h) * SEQ + i) * DH + dh)] = (bf16_t)(val * 0.125f);
                } else if (nb < 32) {
                    const unsigned c2 = (unsigned)col - DIMM;
                    const unsigned h = c2 >> 6, dh = c2 & 63;
                    Kb[(((b * HEADS + h) * SEQ + i) * DH + dh)] = (bf16_t)val;
                } else {
                    const unsigned c2 = (unsigned)col - 2 * DIMM;
                    const unsigned h = c2 >> 6, dh = c2 & 63;
                    Vt[(((b * HEADS + h) * DH + dh) * SEQ + i)] = (bf16_t)val;
                }
            }
        }
    }
}

// ---------------------------------------------------------------------------
// 5) Flash attention: one wave per (b,h,16-query tile), online softmax over
//    32-key chunks.  All K/V fragments batched ahead of their WMMAs; next
//    chunk prefetched during the softmax phase.
// ---------------------------------------------------------------------------
__global__ void attn_kernel(const bf16_t* __restrict__ Qb,
                            const bf16_t* __restrict__ Kb,
                            const bf16_t* __restrict__ Vt,
                            const float*  __restrict__ pb_t,
                            const int*    __restrict__ amask,
                            bf16_t* __restrict__ ao) {
    const int wv = threadIdx.x >> 5;
    const int wid = __builtin_amdgcn_readfirstlane(
        blockIdx.x * (blockDim.x >> 5) + wv);
    const int lane = threadIdx.x & 31;
    const int ml = lane & 15, gr = lane >> 4;
    const int qt = wid & 127;             // 128 query tiles
    const int h = (wid >> 7) & 15;
    const int b = wid >> 11;
    const int ibase = qt * 16;
    const int bh = b * HEADS + h;

    // wave-uniform (scalar) bases
    const bf16_t* __restrict__ Qbase = Qb + (size_t)bh * SEQ * DH + (size_t)ibase * DH;
    const bf16_t* __restrict__ Kbase = Kb + (size_t)bh * SEQ * DH;
    const bf16_t* __restrict__ Vbase = Vt + (size_t)bh * DH * SEQ;
    const int* __restrict__ mrow = amask + b * SEQ;
    bf16_t* __restrict__ aob = ao + (size_t)b * SEQ * DIMM + (size_t)ibase * DIMM + h * DH;

    Frag qf[2];
#pragma unroll
    for (int c = 0; c < 2; ++c)
        load_afrag(qf[c], Qbase, (unsigned)ml * DH + c * 32 + gr * 8);

    v8f o[4] = {};
    float m_run[8], l_run[8];
#pragma unroll
    for (int r = 0; r < 8; ++r) { m_run[r] = -1e30f; l_run[r] = 0.0f; }

    __shared__ __align__(32) bf16_t plds[8][16 * 32];   // per-wave P scratch
    bf16_t* pw = plds[wv];

    for (int kb2 = 0; kb2 < SEQ; kb2 += 32) {
        // ---- load all K fragments for this 32-key chunk ----
        Frag kf[2][2];                     // [n-subtile][dh-chunk]
#pragma unroll
        for (int nt = 0; nt < 2; ++nt) {
            const unsigned ko = (unsigned)(kb2 + nt * 16 + ml) * DH + gr * 8;
            load_bfrag(kf[nt][0], Kbase, ko);
            load_bfrag(kf[nt][1], Kbase, ko + 32);
        }
        // ---- scores: S = (Q*0.125) @ K^T ----
        v8f sAcc[2];
#pragma unroll
        for (int nt = 0; nt < 2; ++nt) {
            v8f s = {};
            s = wmma_bf16f32(qf[0].v, kf[nt][0].v, s);
            s = wmma_bf16f32(qf[1].v, kf[nt][1].v, s);
            sAcc[nt] = s;
        }
        // ---- prefetch next chunk's K/V while doing softmax (VALU phase) ----
        if (kb2 + 32 < SEQ) {
            __builtin_prefetch(Kbase + (unsigned)(kb2 + 32 + ml) * DH, 0, 0);
            __builtin_prefetch(Kbase + (unsigned)(kb2 + 48 + ml) * DH, 0, 0);
            __builtin_prefetch(Vbase + (unsigned)ml * SEQ + kb2 + 32, 0, 0);
            __builtin_prefetch(Vbase + (unsigned)(48 + ml) * SEQ + kb2 + 32, 0, 0);
        }
        // ---- bias + mask + online softmax ----
        const int j0 = kb2 + ml;
        const int j1 = kb2 + 16 + ml;
        const int mok0 = mrow[j0];
        const int mok1 = mrow[j1];
#pragma unroll
        for (int r = 0; r < 8; ++r) {
            const int i = ibase + r + 8 * gr;
            const float bv0 = mok0 ? pb_t[(unsigned)(i - j0 + SEQ - 1) * HEADS + h] : -1e30f;
            const float bv1 = mok1 ? pb_t[(unsigned)(i - j1 + SEQ - 1) * HEADS + h] : -1e30f;
            const float sv0 = sAcc[0][r] + bv0;
            const float sv1 = sAcc[1][r] + bv1;
            float tm = fmaxf(sv0, sv1);
#pragma unroll
            for (int msk = 1; msk < 16; msk <<= 1) tm = fmaxf(tm, __shfl_xor(tm, msk, 32));
            const float mn = fmaxf(m_run[r], tm);
            const float sc = __expf(m_run[r] - mn);
            const float p0 = __expf(sv0 - mn);
            const float p1 = __expf(sv1 - mn);
            float rs = p0 + p1;
#pragma unroll
            for (int msk = 1; msk < 16; msk <<= 1) rs += __shfl_xor(rs, msk, 32);
            l_run[r] = l_run[r] * sc + rs;
            m_run[r] = mn;
#pragma unroll
            for (int nd = 0; nd < 4; ++nd) o[nd][r] *= sc;
            pw[(r + 8 * gr) * 32 + ml]      = (bf16_t)p0;   // D-layout -> LDS
            pw[(r + 8 * gr) * 32 + 16 + ml] = (bf16_t)p1;
        }
        asm volatile("s_wait_dscnt 0" ::: "memory");
        // ---- reload P in A-fragment layout; load all V fragments ----
        Frag pf;
        pf.q[0] = *reinterpret_cast<const uint4*>(pw + ml * 32 + gr * 8);
        pf.q[1] = *reinterpret_cast<const uint4*>(pw + ml * 32 + gr * 8 + 16);
        Frag vf[4];
#pragma unroll
        for (int nd = 0; nd < 4; ++nd)
            load_bfrag(vf[nd], Vbase, (unsigned)(nd * 16 + ml) * SEQ + kb2 + gr * 8);
        // ---- O += P @ V ----
#pragma unroll
        for (int nd = 0; nd < 4; ++nd)
            o[nd] = wmma_bf16f32(pf.v, vf[nd].v, o[nd]);
    }
    // ---- normalize and write [b,n,dim] bf16 ----
    float linv[8];
#pragma unroll
    for (int r = 0; r < 8; ++r) linv[r] = 1.0f / l_run[r];
#pragma unroll
    for (int nd = 0; nd < 4; ++nd) {
#pragma unroll
        for (int r = 0; r < 8; ++r) {
            const float val = o[nd][r] * linv[r];
            aob[(unsigned)(r + 8 * gr) * DIMM + nd * 16 + ml] = (bf16_t)val;
        }
    }
}

// ---------------------------------------------------------------------------
// 6) Output projection: [4096,1024]bf16 x [1024,1024]bf16 + b_out -> f32.
//    Same 32x64 double-buffered wave tile.
// ---------------------------------------------------------------------------
__global__ void out_gemm_kernel(const bf16_t* __restrict__ A,
                                const bf16_t* __restrict__ Bt,   // [1024][1024]
                                const float* __restrict__ bias,
                                float* __restrict__ out) {
    const int wid = __builtin_amdgcn_readfirstlane(
        blockIdx.x * (blockDim.x >> 5) + (threadIdx.x >> 5));
    const int lane = threadIdx.x & 31;
    const int ml = lane & 15, gr = lane >> 4;
    const int mt = wid & 127;        // 128 M-tiles of 32 rows
    const int nb = wid >> 7;         // 16 N-tiles of 64

    const unsigned aoff0 = (unsigned)(mt * 32 + ml) * DIMM + gr * 8;
    const unsigned aoff1 = aoff0 + 16u * DIMM;
    unsigned boff[4];
#pragma unroll
    for (int nd = 0; nd < 4; ++nd)
        boff[nd] = (unsigned)(nb * 64 + nd * 16 + ml) * DIMM + gr * 8;

    v8f acc[2][4] = {};
    Frag af[2][2];
    Frag bfm[2][4];
    load_afrag(af[0][0], A, aoff0);
    load_afrag(af[0][1], A, aoff1);
#pragma unroll
    for (int nd = 0; nd < 4; ++nd) load_bfrag(bfm[0][nd], Bt, boff[nd]);

    for (int kk = 0; kk < DIMM; kk += 32) {
        const int cur = (kk >> 5) & 1;
        const int nxt = cur ^ 1;
        if (kk + 32 < DIMM) {
            load_afrag(af[nxt][0], A, aoff0 + kk + 32);
            load_afrag(af[nxt][1], A, aoff1 + kk + 32);
#pragma unroll
            for (int nd = 0; nd < 4; ++nd) load_bfrag(bfm[nxt][nd], Bt, boff[nd] + kk + 32);
        }
#pragma unroll
        for (int s = 0; s < 2; ++s)
#pragma unroll
            for (int nd = 0; nd < 4; ++nd)
                acc[s][nd] = wmma_bf16f32(af[cur][s].v, bfm[cur][nd].v, acc[s][nd]);
    }

#pragma unroll
    for (int s = 0; s < 2; ++s) {
#pragma unroll
        for (int nd = 0; nd < 4; ++nd) {
            const int col = nb * 64 + nd * 16 + ml;
            const float bo = bias[col];
#pragma unroll
            for (int r = 0; r < 8; ++r) {
                const unsigned row = (unsigned)(mt * 32 + s * 16 + r + 8 * gr);
                out[row * DIMM + col] = acc[s][nd][r] + bo;
            }
        }
    }
}

// ---------------------------------------------------------------------------
// Launch
// ---------------------------------------------------------------------------
extern "C" void kernel_launch(void* const* d_in, const int* in_sizes, int n_in,
                              void* d_out, int out_size, void* d_ws, size_t ws_size,
                              hipStream_t stream) {
    const float* x      = (const float*)d_in[0];
    const int*   amask  = (const int*)  d_in[1];
    const float* g      = (const float*)d_in[2];
    const float* w_qkv  = (const float*)d_in[3];
    const float* w_out  = (const float*)d_in[4];
    const float* b_out  = (const float*)d_in[5];
    const float* pb_w1  = (const float*)d_in[6];
    const float* pb_b1  = (const float*)d_in[7];
    const float* pb_w2  = (const float*)d_in[8];
    const float* pb_b2  = (const float*)d_in[9];
    const float* pb_w3  = (const float*)d_in[10];
    const float* pb_b3  = (const float*)d_in[11];
    float* out = (float*)d_out;

    char* ws = (char*)d_ws;
    const size_t ROWS = (size_t)BATCH * SEQ;                 // 4096
    size_t off = 0;
    bf16_t* xn    = (bf16_t*)(ws + off); off += ROWS * DIMM * 2;            // 8 MB
    bf16_t* wqkvT = (bf16_t*)(ws + off); off += (size_t)3 * DIMM * DIMM * 2;// 6 MB
    bf16_t* woutT = (bf16_t*)(ws + off); off += (size_t)DIMM * DIMM * 2;    // 2 MB
    bf16_t* Qb    = (bf16_t*)(ws + off); off += ROWS * DIMM * 2;            // 8 MB
    bf16_t* Kb    = (bf16_t*)(ws + off); off += ROWS * DIMM * 2;            // 8 MB
    bf16_t* Vt    = (bf16_t*)(ws + off); off += ROWS * DIMM * 2;            // 8 MB
    bf16_t* ao    = (bf16_t*)(ws + off); off += ROWS * DIMM * 2;            // 8 MB
    float*  pbt   = (float*) (ws + off); off += (size_t)NREL * HEADS * 4;   // 256 KB
    (void)ws_size; (void)in_sizes; (void)n_in; (void)out_size;

    rmsnorm_kernel<<<(int)ROWS, 256, 0, stream>>>(x, g, xn);
    conv_transpose_kernel<<<(3 * DIMM * DIMM) / 256, 256, 0, stream>>>(w_qkv, wqkvT, DIMM, 3 * DIMM);
    conv_transpose_kernel<<<(DIMM * DIMM) / 256, 256, 0, stream>>>(w_out, woutT, DIMM, DIMM);
    pb_mlp_kernel<<<NREL, 256, 0, stream>>>(pb_w1, pb_b1, pb_w2, pb_b2, pb_w3, pb_b3, pbt);
    // 128 M-tiles x 48 N-tiles = 6144 waves, 8 waves/block
    qkv_gemm_kernel<<<6144 / 8, 256, 0, stream>>>(xn, wqkvT, Qb, Kb, Vt);
    // 2*16*128 = 4096 waves
    attn_kernel<<<4096 / 8, 256, 0, stream>>>(Qb, Kb, Vt, pbt, amask, ao);
    // 128 M-tiles x 16 N-tiles = 2048 waves
    out_gemm_kernel<<<2048 / 8, 256, 0, stream>>>(ao, woutT, b_out, out);
}
